// Attention_Block_9423158247686
// MI455X (gfx1250) — matrix operand order repaired and checked by reading, not compiled, not run on hardware
//
#include <hip/hip_runtime.h>

// ---------------------------------------------------------------------------
// Attention block for MI455X (gfx1250, wave32, WMMA 16x16x32 bf16).
// Pipeline: LN -> (Q,K,V) bf16 GEMMs -> flash attention (online softmax)
//           -> output projection (f32 out).
// ---------------------------------------------------------------------------

typedef __bf16  v16bf __attribute__((ext_vector_type(16)));
typedef __bf16  v8bf  __attribute__((ext_vector_type(8)));
typedef __bf16  v4bf  __attribute__((ext_vector_type(4)));
typedef float   v8f   __attribute__((ext_vector_type(8)));

#define BATCH 4
#define SEQ   2048
#define DIM   512
#define HEADS 8
#define DK    64
#define ROWS  (BATCH * SEQ)          // 8192

// ---------------------------------------------------------------------------
// Kernel 1: LayerNorm, f32 -> bf16.  One wave per 512-wide row.
// grid = ROWS/8 blocks of 256 threads (8 waves).
// ---------------------------------------------------------------------------
__global__ __launch_bounds__(256) void ln_kernel(
    const float* __restrict__ x, const float* __restrict__ gamma,
    const float* __restrict__ beta, __bf16* __restrict__ xn)
{
    const int lane = threadIdx.x & 31;
    const int wave = threadIdx.x >> 5;
    const int row  = blockIdx.x * 8 + wave;

    const float4* xr = reinterpret_cast<const float4*>(x + (size_t)row * DIM);
    float4 v[4];
    float s = 0.f, ss = 0.f;
#pragma unroll
    for (int i = 0; i < 4; ++i) {
        v[i] = xr[i * 32 + lane];
        s  += v[i].x + v[i].y + v[i].z + v[i].w;
        ss += v[i].x*v[i].x + v[i].y*v[i].y + v[i].z*v[i].z + v[i].w*v[i].w;
    }
#pragma unroll
    for (int off = 1; off < 32; off <<= 1) {
        s  += __shfl_xor(s,  off, 32);
        ss += __shfl_xor(ss, off, 32);
    }
    const float mu   = s * (1.0f / DIM);
    const float var  = ss * (1.0f / DIM) - mu * mu;
    const float rstd = rsqrtf(var + 1e-5f);

    const float4* g4 = reinterpret_cast<const float4*>(gamma);
    const float4* b4 = reinterpret_cast<const float4*>(beta);
#pragma unroll
    for (int i = 0; i < 4; ++i) {
        const int idx = i * 32 + lane;          // float4 index within row
        float4 g = g4[idx], b = b4[idx];
        v4bf o;
        o[0] = (__bf16)((v[i].x - mu) * rstd * g.x + b.x);
        o[1] = (__bf16)((v[i].y - mu) * rstd * g.y + b.y);
        o[2] = (__bf16)((v[i].z - mu) * rstd * g.z + b.z);
        o[3] = (__bf16)((v[i].w - mu) * rstd * g.w + b.w);
        *reinterpret_cast<v4bf*>(xn + (size_t)row * DIM + idx * 4) = o;
    }
}

// ---------------------------------------------------------------------------
// Kernel 2: GEMM  C[M=8192, N=512] = A_bf16 @ W_f32^T (+bias)
// Block tile 128(M) x 64(N), K staged 32 at a time in LDS.
// 8 waves; wave w owns rows [16w,16w+16) -> 4 wmma accumulators (16x64).
// mode 0: bf16 out, split-head (b,h,n,dk)      (Q, V)
// mode 1: bf16 out, split-head transposed (b,h,dk,n)  (K^T)
// mode 2: f32 out, (b,n,dim) directly          (final projection)
// ---------------------------------------------------------------------------
__global__ __launch_bounds__(256) void gemm_kernel(
    const __bf16* __restrict__ A, const float* __restrict__ W,
    const float* __restrict__ bias,
    __bf16* __restrict__ outb, float* __restrict__ outf, int mode)
{
    __shared__ __bf16 lA[128][40];   // [m][k] pad to 40
    __shared__ __bf16 lB[32][72];    // [k][n] pad to 72

    const int t     = threadIdx.x;
    const int lane  = t & 31;
    const int w     = t >> 5;
    const int half  = lane >> 4;
    const int l15   = lane & 15;
    const int mbase = blockIdx.x * 128;
    const int n0    = blockIdx.y * 64;

    v8f acc[4] = {};

    for (int k0 = 0; k0 < DIM; k0 += 32) {
        // ---- stage A tile: 128 x 32 bf16 (each thread: one 16-elem chunk)
        {
            const int r  = t >> 1;
            const int kc = (t & 1) * 16;
            const v8bf* src = reinterpret_cast<const v8bf*>(
                A + (size_t)(mbase + r) * DIM + k0 + kc);
            v8bf a0 = src[0];
            v8bf a1 = src[1];
            *reinterpret_cast<v8bf*>(&lA[r][kc])     = a0;
            *reinterpret_cast<v8bf*>(&lA[r][kc + 8]) = a1;
        }
        // ---- stage B tile: W[n0..n0+63][k0..k0+31] f32 -> bf16, transposed
        {
            const int n  = t >> 2;
            const int kc = (t & 3) * 8;
            const float4* src = reinterpret_cast<const float4*>(
                W + (size_t)(n0 + n) * DIM + k0 + kc);
            float4 w0 = src[0], w1 = src[1];
            lB[kc + 0][n] = (__bf16)w0.x;  lB[kc + 1][n] = (__bf16)w0.y;
            lB[kc + 2][n] = (__bf16)w0.z;  lB[kc + 3][n] = (__bf16)w0.w;
            lB[kc + 4][n] = (__bf16)w1.x;  lB[kc + 5][n] = (__bf16)w1.y;
            lB[kc + 6][n] = (__bf16)w1.z;  lB[kc + 7][n] = (__bf16)w1.w;
        }
        __syncthreads();

        // ---- A fragment: row = 16*w + l15; K runs {half*8+0..7, 16+half*8+0..7}
        const int mrow = (w << 4) + l15;
        v8bf alo = *reinterpret_cast<const v8bf*>(&lA[mrow][half * 8]);
        v8bf ahi = *reinterpret_cast<const v8bf*>(&lA[mrow][16 + half * 8]);
        v16bf af;
#pragma unroll
        for (int i = 0; i < 8; ++i) { af[i] = alo[i]; af[i + 8] = ahi[i]; }

#pragma unroll
        for (int nt = 0; nt < 4; ++nt) {
            // B fragment: col N = l15; K runs {half*8+0..7, 16+half*8+0..7}
            v16bf bfr;
#pragma unroll
            for (int i = 0; i < 8; ++i) { bfr[i] = lB[half * 8 + i][nt * 16 + l15]; bfr[i + 8] = lB[16 + half * 8 + i][nt * 16 + l15]; }
            acc[nt] = __builtin_amdgcn_wmma_f32_16x16x32_bf16(
                false, af, false, bfr, (short)0, acc[nt], false, false);
        }
        __syncthreads();
    }

    // ---- epilogue: bias + store per layout mode
#pragma unroll
    for (int nt = 0; nt < 4; ++nt) {
        const int o  = n0 + nt * 16 + l15;
        const float bv = bias[o];
#pragma unroll
        for (int e = 0; e < 8; ++e) {
            const float val = acc[nt][e] + bv;
            const int mg = mbase + (w << 4) + e + 8 * half;   // global row
            if (mode == 2) {
                outf[(size_t)mg * DIM + o] = val;
            } else {
                const int b = mg >> 11, n = mg & 2047;
                const int h = o >> 6,  d = o & 63;
                size_t idx;
                if (mode == 0)
                    idx = ((size_t)(b * HEADS + h) * SEQ + n) * DK + d;
                else
                    idx = ((size_t)(b * HEADS + h) * DK + d) * SEQ + n;
                outb[idx] = (__bf16)val;
            }
        }
    }
}

// ---------------------------------------------------------------------------
// Kernel 3: flash attention. One wave per 16 queries, dk = 64.
// Q frags live in registers; loop over keys in chunks of 32:
//   scores (2x2 wmma) -> online softmax -> P via LDS relayout -> PV (4 wmma).
// grid = (SEQ/128, BATCH*HEADS), block = 256 (8 waves).
// ---------------------------------------------------------------------------
__global__ __launch_bounds__(256) void attn_kernel(
    const __bf16* __restrict__ Q, const __bf16* __restrict__ KT,
    const __bf16* __restrict__ V, __bf16* __restrict__ AO)
{
    __shared__ __bf16 lP[8][16][40];   // per-wave P staging (C-layout -> A-layout)

    const int t    = threadIdx.x;
    const int lane = t & 31;
    const int w    = t >> 5;
    const int half = lane >> 4;
    const int l15  = lane & 15;
    const int bh   = blockIdx.y;
    const int qbase = blockIdx.x * 128 + w * 16;

    const __bf16* qp = Q  + (size_t)bh * SEQ * DK;
    const __bf16* kp = KT + (size_t)bh * DK * SEQ;
    const __bf16* vp = V  + (size_t)bh * SEQ * DK;

    // Q fragments: A 16x32 for d in [0,32) and [32,64)
    v16bf qf[2];
#pragma unroll
    for (int ks = 0; ks < 2; ++ks) {
        const __bf16* base = qp + (size_t)(qbase + l15) * DK + ks * 32 + half * 8;
        v8bf a0 = *reinterpret_cast<const v8bf*>(base);
        v8bf a1 = *reinterpret_cast<const v8bf*>(base + 16);
#pragma unroll
        for (int i = 0; i < 8; ++i) { qf[ks][i] = a0[i]; qf[ks][i + 8] = a1[i]; }
    }

    float Mr[8], Lr[8];
#pragma unroll
    for (int e = 0; e < 8; ++e) { Mr[e] = -1e30f; Lr[e] = 0.f; }
    v8f O[4] = {};

    for (int m0 = 0; m0 < SEQ; m0 += 32) {
        // ---- scores S = Q K^T (16 x 32), two 16x16 tiles
        v8f s[2];
#pragma unroll
        for (int nt = 0; nt < 2; ++nt) {
            v8f c = {};
#pragma unroll
            for (int ks = 0; ks < 2; ++ks) {
                const __bf16* base = kp + (size_t)(ks * 32 + half * 8) * SEQ + m0 + nt * 16 + l15;
                v16bf bfr;
#pragma unroll
                for (int i = 0; i < 8; ++i) { bfr[i] = base[(size_t)i * SEQ]; bfr[i + 8] = base[(size_t)(16 + i) * SEQ]; }
                c = __builtin_amdgcn_wmma_f32_16x16x32_bf16(
                    false, qf[ks], false, bfr, (short)0, c, false, false);
            }
            s[nt] = c;
        }

        // ---- online softmax (rows live at (e, half); cols across 16 lanes)
        float alpha[8];
#pragma unroll
        for (int e = 0; e < 8; ++e) {
            float s0 = s[0][e] * 0.125f;     // 1/sqrt(64)
            float s1 = s[1][e] * 0.125f;
            float mx = fmaxf(s0, s1);
#pragma unroll
            for (int off = 1; off < 16; off <<= 1)
                mx = fmaxf(mx, __shfl_xor(mx, off, 32));
            const float mnew = fmaxf(Mr[e], mx);
            const float a = __expf(Mr[e] - mnew);
            Mr[e] = mnew; alpha[e] = a;
            const float p0 = __expf(s0 - mnew);
            const float p1 = __expf(s1 - mnew);
            s[0][e] = p0; s[1][e] = p1;
            float rs = p0 + p1;
#pragma unroll
            for (int off = 1; off < 16; off <<= 1)
                rs += __shfl_xor(rs, off, 32);
            Lr[e] = Lr[e] * a + rs;
        }
#pragma unroll
        for (int nt = 0; nt < 4; ++nt)
#pragma unroll
            for (int e = 0; e < 8; ++e) O[nt][e] *= alpha[e];

        // ---- P: C-layout -> LDS -> A-layout fragment (in-order DS per wave)
#pragma unroll
        for (int e = 0; e < 8; ++e) {
            lP[w][e + 8 * half][l15]      = (__bf16)s[0][e];
            lP[w][e + 8 * half][16 + l15] = (__bf16)s[1][e];
        }
        v16bf pf;
        {
            const __bf16* base = &lP[w][l15][half * 8];
            v8bf a0 = *reinterpret_cast<const v8bf*>(base);
            v8bf a1 = *reinterpret_cast<const v8bf*>(base + 16);
#pragma unroll
            for (int i = 0; i < 8; ++i) { pf[i] = a0[i]; pf[i + 8] = a1[i]; }
        }

        // ---- O += P @ V  (K = 32 keys, N = 64 dk as 4 tiles)
#pragma unroll
        for (int nt = 0; nt < 4; ++nt) {
            const __bf16* base = vp + (size_t)(m0 + half * 8) * DK + nt * 16 + l15;
            v16bf vf;
#pragma unroll
            for (int i = 0; i < 8; ++i) { vf[i] = base[i * DK]; vf[i + 8] = base[(16 + i) * DK]; }
            O[nt] = __builtin_amdgcn_wmma_f32_16x16x32_bf16(
                false, pf, false, vf, (short)0, O[nt], false, false);
        }
    }

    // ---- normalize and store to (b, n, h*dk) bf16
    const int b = bh >> 3, h = bh & 7;
#pragma unroll
    for (int e = 0; e < 8; ++e) {
        const float inv = 1.0f / Lr[e];
        const int nrow = qbase + e + 8 * half;
        const size_t rowoff = ((size_t)(b * SEQ + nrow)) * DIM + h * DK;
#pragma unroll
        for (int nt = 0; nt < 4; ++nt)
            AO[rowoff + nt * 16 + l15] = (__bf16)(O[nt][e] * inv);
    }
}

// ---------------------------------------------------------------------------
extern "C" void kernel_launch(void* const* d_in, const int* in_sizes, int n_in,
                              void* d_out, int out_size, void* d_ws, size_t ws_size,
                              hipStream_t stream) {
    const float* x     = (const float*)d_in[0];
    const float* gamma = (const float*)d_in[1];
    const float* beta  = (const float*)d_in[2];
    const float* Wq    = (const float*)d_in[3];
    const float* bq    = (const float*)d_in[4];
    const float* Wk    = (const float*)d_in[5];
    const float* bk    = (const float*)d_in[6];
    const float* Wv    = (const float*)d_in[7];
    const float* bv    = (const float*)d_in[8];
    const float* Wo    = (const float*)d_in[9];
    const float* bo    = (const float*)d_in[10];
    float* out = (float*)d_out;

    // workspace: 5 bf16 buffers of ROWS*DIM = 4 Mi elements each (40 MB total)
    const size_t SZ = (size_t)ROWS * DIM;
    __bf16* ws  = (__bf16*)d_ws;
    __bf16* xn  = ws;            // LN output       (b, n, dim)
    __bf16* qb  = ws + 1 * SZ;   // Q               (b, h, n, dk)
    __bf16* kTb = ws + 2 * SZ;   // K transposed    (b, h, dk, n)
    __bf16* vb  = ws + 3 * SZ;   // V               (b, h, n, dk)
    __bf16* ao  = ws + 4 * SZ;   // attention out   (b, n, dim)

    ln_kernel<<<ROWS / 8, 256, 0, stream>>>(x, gamma, beta, xn);

    dim3 ggrid(ROWS / 128, DIM / 64);
    gemm_kernel<<<ggrid, 256, 0, stream>>>(xn, Wq, bq, qb,  nullptr, 0);
    gemm_kernel<<<ggrid, 256, 0, stream>>>(xn, Wk, bk, kTb, nullptr, 1);
    gemm_kernel<<<ggrid, 256, 0, stream>>>(xn, Wv, bv, vb,  nullptr, 0);

    attn_kernel<<<dim3(SEQ / 128, BATCH * HEADS), 256, 0, stream>>>(qb, kTb, vb, ao);

    gemm_kernel<<<ggrid, 256, 0, stream>>>(ao, Wo, bo, nullptr, out, 2);
}
